// GINELayer_83004537962843
// MI455X (gfx1250) — matrix operand order repaired and checked by reading, not compiled, not run on hardware
//
#include <hip/hip_runtime.h>

// ---------------------------------------------------------------------------
// Problem constants (from reference): N=50000 nodes, E=400000 edges,
// H=512 hidden, ED=64 edge-dim.
// ---------------------------------------------------------------------------
static constexpr int cN  = 50000;
static constexpr int cE  = 400000;
static constexpr int cH  = 512;
static constexpr int cED = 64;
static constexpr float BN_EPS = 1e-5f;

typedef __attribute__((ext_vector_type(16))) __bf16        v16bf;
typedef __attribute__((ext_vector_type(8)))  float         v8f;
typedef __attribute__((ext_vector_type(4)))  unsigned int  u32x4;
typedef __attribute__((ext_vector_type(4)))  int           i32x4;
typedef __attribute__((ext_vector_type(8)))  int           i32x8;

union BF16Frag {
  v16bf          v;
  u32x4          q[2];
  unsigned short h[16];
};

__device__ __forceinline__ unsigned short f2bf(float f) {
  unsigned int u = __float_as_uint(f);
  u += 0x7FFFu + ((u >> 16) & 1u);   // round-to-nearest-even
  return (unsigned short)(u >> 16);
}

// ---------------------------------------------------------------------------
// Tensor Data Mover: stage a [rows][K] bf16 tile (row-major, contiguous rows)
// from global memory into LDS.  D# per CDNA5 ISA ch.8:
//   group0: count=1, lds_addr, global_addr, type=2
//   group1: data_size=2B, tensor_dim0=K, tensor_dim1=rows, tile_dim0=K,
//           tile_dim1=rows, tensor_dim0_stride=K
// One wave issues the DMA; it is tracked with TENSORcnt.
// This toolchain exposes the 6-arg builtin:
//   (u32x4 g0, i32x8 g1, i32x4 g2, i32x4 g3, i32x8 pad, i32 cpol)
// ---------------------------------------------------------------------------
#if __has_builtin(__builtin_amdgcn_tensor_load_to_lds)
#define USE_TDM 1
#else
#define USE_TDM 0
#endif

#if USE_TDM
__device__ __forceinline__ void tdm_load_tile_bf16(const unsigned short* gbase,
                                                   unsigned int lds_off_bytes,
                                                   int K, int rows) {
  const unsigned long long ga = (unsigned long long)(uintptr_t)gbase;
  u32x4 g0;
  g0[0] = 1u;                                            // count=1 (valid), user mode
  g0[1] = lds_off_bytes;                                 // lds_addr
  g0[2] = (unsigned)(ga & 0xFFFFFFFFu);                  // global_addr[31:0]
  g0[3] = (unsigned)((ga >> 32) & 0x01FFFFFFu) | (2u << 30);  // global_addr[56:32], type=2
  i32x8 g1;
  g1[0] = (int)(1u << 16);                               // wg_mask=0, data_size=1 (2 bytes)
  g1[1] = (int)(((unsigned)K & 0xFFFFu) << 16);          // tensor_dim0[15:0]  @ bits 63:48
  g1[2] = (int)((((unsigned)K >> 16) & 0xFFFFu) |        // tensor_dim0[31:16] @ bits 79:64
                (((unsigned)rows & 0xFFFFu) << 16));     // tensor_dim1[15:0]  @ bits 95:80
  g1[3] = (int)(((((unsigned)rows >> 16) & 0xFFFFu)) |   // tensor_dim1[31:16] @ bits 111:96
                (((unsigned)K & 0xFFFFu) << 16));        // tile_dim0          @ bits 127:112
  g1[4] = (int)((unsigned)rows & 0xFFFFu);               // tile_dim1; tile_dim2=0 (2D)
  g1[5] = (int)(unsigned)K;                              // tensor_dim0_stride[31:0]
  g1[6] = 0;                                             // stride hi / dim1_stride lo
  g1[7] = 0;
  const i32x4 gz4 = {0, 0, 0, 0};
  const i32x8 gz8 = {0, 0, 0, 0, 0, 0, 0, 0};
  __builtin_amdgcn_tensor_load_to_lds(g0, g1, gz4, gz4, gz8, 0);
  __builtin_amdgcn_s_wait_tensorcnt(0);
}
#endif

// ---------------------------------------------------------------------------
// Generic helpers
// ---------------------------------------------------------------------------
__global__ void k_zero(float* __restrict__ p, long long n) {
  long long i = (long long)blockIdx.x * blockDim.x + threadIdx.x;
  long long stride = (long long)gridDim.x * blockDim.x;
  for (; i < n; i += stride) p[i] = 0.0f;
}

// W [K][Ncols] f32 row-major  ->  WT [Ncols][K] bf16 (transposed)
__global__ void k_convT(const float* __restrict__ W, unsigned short* __restrict__ WT,
                        int K, int Ncols) {
  long long total = (long long)K * Ncols;
  long long i = (long long)blockIdx.x * blockDim.x + threadIdx.x;
  long long stride = (long long)gridDim.x * blockDim.x;
  for (; i < total; i += stride) {
    int k = (int)(i / Ncols), n = (int)(i % Ncols);
    WT[(size_t)n * K + k] = f2bf(W[i]);
  }
}

// ---------------------------------------------------------------------------
// Edge stage: lin_e = edge_attr @ W_e (bf16 WMMA), msg = relu(x[src]+lin_e+b_e),
// scatter-add into aggr[dst].  One block = 16 edges x 512 cols; 8 waves,
// each wave owns 64 columns (4 n-tiles), K=64 -> 2 WMMA steps per tile.
// ---------------------------------------------------------------------------
__global__ __launch_bounds__(256) void k_edge(
    const float* __restrict__ x, const float* __restrict__ ea,
    const int* __restrict__ ei, const unsigned short* __restrict__ WeT,
    const float* __restrict__ be, float* __restrict__ aggr) {
  __shared__ unsigned short As[16 * 64];
  const int tid   = threadIdx.x;
  const int ebase = blockIdx.x * 16;

  // Stage 16x64 edge_attr tile into LDS as bf16
  for (int i = tid; i < 16 * 64; i += 256)
    As[i] = f2bf(ea[(size_t)ebase * 64 + i]);
  __syncthreads();

  const int wave = tid >> 5, lane = tid & 31;
  const int half = lane >> 4, l15 = lane & 15;

  // A fragments, contiguous k order (matches B):
  // lanes 0-15 (row=lane):  K = kb+0..15
  // lanes 16-31 (row=lane-16): K = kb+16..31
  BF16Frag a0, a1;
  {
    const unsigned short* p = &As[l15 * 64];
#pragma unroll
    for (int j = 0; j < 8; ++j) {
      a0.h[j]     = p[half * 16 + j];
      a0.h[8 + j] = p[half * 16 + 8 + j];
      a1.h[j]     = p[32 + half * 16 + j];
      a1.h[8 + j] = p[32 + half * 16 + 8 + j];
    }
  }

  // C-tile rows this lane touches: M = r + 8*half.  Hoist row pointers.
  const float* xrow[8];
  float*       arow[8];
#pragma unroll
  for (int r = 0; r < 8; ++r) {
    const int e = ebase + half * 8 + r;
    xrow[r] = x    + (size_t)ei[e]      * cH;
    arow[r] = aggr + (size_t)ei[cE + e] * cH;
  }

  const v8f vz = {0.f, 0.f, 0.f, 0.f, 0.f, 0.f, 0.f, 0.f};
#pragma unroll
  for (int i = 0; i < 4; ++i) {
    const int col = (wave * 4 + i) * 16 + l15;
    // B fragments: lanes 0-15 K=kb..kb+15, lanes 16-31 K=kb+16..kb+31,
    // contiguous in transposed WeT[col][k]
    BF16Frag b0, b1;
    const u32x4* pb0 = (const u32x4*)(WeT + (size_t)col * 64 + half * 16);
    b0.q[0] = pb0[0]; b0.q[1] = pb0[1];
    const u32x4* pb1 = (const u32x4*)(WeT + (size_t)col * 64 + 32 + half * 16);
    b1.q[0] = pb1[0]; b1.q[1] = pb1[1];

    v8f acc = __builtin_amdgcn_wmma_f32_16x16x32_bf16(false, a0.v, false, b0.v,
                                                      (short)0, vz, false, false);
    acc = __builtin_amdgcn_wmma_f32_16x16x32_bf16(false, a1.v, false, b1.v,
                                                  (short)0, acc, false, false);
    const float bc = be[col];
#pragma unroll
    for (int r = 0; r < 8; ++r) {
      float v = acc[r] + bc + xrow[r][col];
      v = v > 0.f ? v : 0.f;
      atomicAdd(&arow[r][col], v);
    }
  }
}

// ---------------------------------------------------------------------------
// Combine: hbf = bf16((1+eps)*x + aggr)
// ---------------------------------------------------------------------------
__global__ void k_combine(const float* __restrict__ x, const float* __restrict__ aggr,
                          const float* __restrict__ epsp, unsigned short* __restrict__ hbf,
                          long long n) {
  const float e1 = 1.0f + epsp[0];
  long long i = (long long)blockIdx.x * blockDim.x + threadIdx.x;
  long long stride = (long long)gridDim.x * blockDim.x;
  for (; i < n; i += stride) hbf[i] = f2bf(e1 * x[i] + aggr[i]);
}

// ---------------------------------------------------------------------------
// bf16 GEMM with fused bias + per-column sum/sumsq atomics (for BatchNorm).
// C[Nrows,Ncols] = A[Nrows,K](bf16) @ BT[Ncols,K](bf16)^T + bias.
// Block: 256 threads = 8 waves; block tile 64 rows x 128 cols;
// each wave: 16 rows x 64 cols (4 accumulator tiles), K stepped by 32.
// The 128 x K B tile is staged into LDS once per block by the Tensor Data
// Mover (tensor_load_to_lds + s_wait_tensorcnt), then all fragment reads
// come from LDS.
// ---------------------------------------------------------------------------
__global__ __launch_bounds__(256) void k_gemm_bias_stats(
    const unsigned short* __restrict__ A, const unsigned short* __restrict__ BT,
    const float* __restrict__ bias, float* __restrict__ C,
    float* __restrict__ colsum, float* __restrict__ colsq,
    int Nrows, int K, int Ncols) {
  extern __shared__ unsigned short Bs[];   // [128][K] bf16

  const int wave = threadIdx.x >> 5, lane = threadIdx.x & 31;
  const int half = lane >> 4, l15 = lane & 15;
  const int mbase = (blockIdx.x * 4 + (wave & 3)) * 16;
  const int ntb   = blockIdx.y * 8 + (wave >> 2) * 4;

#if USE_TDM
  if (threadIdx.x < 32) {
    tdm_load_tile_bf16(BT + (size_t)blockIdx.y * 128 * K,
                       (unsigned)(uintptr_t)&Bs[0], K, 128);
  }
  __syncthreads();
#else
  // Fallback: cooperative LDS fill with plain vector loads
  {
    const unsigned short* src = BT + (size_t)blockIdx.y * 128 * K;
    const int total4 = 128 * K / 8;                    // u32x4 = 8 bf16
    u32x4* dst4 = (u32x4*)&Bs[0];
    const u32x4* src4 = (const u32x4*)src;
    for (int i = threadIdx.x; i < total4; i += 256) dst4[i] = src4[i];
  }
  __syncthreads();
#endif

  const int  arowi = mbase + l15;
  const bool rowok = arowi < Nrows;
  const unsigned short* ap = A + (size_t)arowi * K;

  const v8f  vz = {0.f, 0.f, 0.f, 0.f, 0.f, 0.f, 0.f, 0.f};
  const u32x4 qz = {0u, 0u, 0u, 0u};
  v8f acc[4];
  acc[0] = vz; acc[1] = vz; acc[2] = vz; acc[3] = vz;

  // Local (in-LDS) column base for this wave's 4 n-tiles
  const int colloc0 = ((wave >> 2) * 4) * 16 + l15;

  for (int kb = 0; kb < K; kb += 32) {
    BF16Frag a;
    if (rowok) {
      a.q[0] = *(const u32x4*)(ap + kb + half * 16);
      a.q[1] = *(const u32x4*)(ap + kb + half * 16 + 8);
    } else {
      a.q[0] = qz; a.q[1] = qz;
    }
#pragma unroll
    for (int i = 0; i < 4; ++i) {
      const int colloc = colloc0 + i * 16;             // 0..127 within LDS tile
      const u32x4* pb = (const u32x4*)(Bs + (size_t)colloc * K + kb + half * 16);
      BF16Frag b;
      b.q[0] = pb[0]; b.q[1] = pb[1];
      acc[i] = __builtin_amdgcn_wmma_f32_16x16x32_bf16(false, a.v, false, b.v,
                                                       (short)0, acc[i], false, false);
    }
  }

#pragma unroll
  for (int i = 0; i < 4; ++i) {
    const int col = (ntb + i) * 16 + l15;
    const float bc = bias[col];
    float s = 0.f, s2 = 0.f;
#pragma unroll
    for (int r = 0; r < 8; ++r) {
      const int rr = mbase + half * 8 + r;      // C row for accumulator lane r
      if (rr < Nrows) {
        const float v = acc[i][r] + bc;
        C[(size_t)rr * Ncols + col] = v;
        s += v; s2 += v * v;
      }
    }
    atomicAdd(&colsum[col], s);
    atomicAdd(&colsq[col], s2);
  }
}

// ---------------------------------------------------------------------------
// BatchNorm statistics: mean / rstd per column (biased variance)
// ---------------------------------------------------------------------------
__global__ void k_bnstats(const float* __restrict__ colsum, const float* __restrict__ colsq,
                          float* __restrict__ mean, float* __restrict__ rstd,
                          int C, float invN) {
  const int c = blockIdx.x * blockDim.x + threadIdx.x;
  if (c < C) {
    const float m = colsum[c] * invN;
    const float v = colsq[c] * invN - m * m;
    mean[c] = m;
    rstd[c] = rsqrtf(v + BN_EPS);
  }
}

// BN + SiLU -> bf16 (feeds next GEMM)
__global__ void k_bnact_bf16(const float* __restrict__ h, const float* __restrict__ mean,
                             const float* __restrict__ rstd, const float* __restrict__ g,
                             const float* __restrict__ beta, unsigned short* __restrict__ out,
                             long long n, int C) {
  long long i = (long long)blockIdx.x * blockDim.x + threadIdx.x;
  long long stride = (long long)gridDim.x * blockDim.x;
  for (; i < n; i += stride) {
    const int c = (int)(i % C);
    const float t = (h[i] - mean[c]) * rstd[c] * g[c] + beta[c];
    const float s = t / (1.0f + __expf(-t));
    out[i] = f2bf(s);
  }
}

// BN + SiLU -> f32 (final output)
__global__ void k_bnact_f32(const float* __restrict__ h, const float* __restrict__ mean,
                            const float* __restrict__ rstd, const float* __restrict__ g,
                            const float* __restrict__ beta, float* __restrict__ out,
                            long long n, int C) {
  long long i = (long long)blockIdx.x * blockDim.x + threadIdx.x;
  long long stride = (long long)gridDim.x * blockDim.x;
  for (; i < n; i += stride) {
    const int c = (int)(i % C);
    const float t = (h[i] - mean[c]) * rstd[c] * g[c] + beta[c];
    out[i] = t / (1.0f + __expf(-t));
  }
}

// ---------------------------------------------------------------------------
// Host-side launch
// ---------------------------------------------------------------------------
extern "C" void kernel_launch(void* const* d_in, const int* in_sizes, int n_in,
                              void* d_out, int out_size, void* d_ws, size_t ws_size,
                              hipStream_t stream) {
  (void)in_sizes; (void)n_in; (void)out_size; (void)ws_size;

  const float* x     = (const float*)d_in[0];
  const float* ea    = (const float*)d_in[1];
  const int*   ei    = (const int*)  d_in[2];
  const float* W_e   = (const float*)d_in[3];
  const float* b_e   = (const float*)d_in[4];
  const float* W1    = (const float*)d_in[5];
  const float* b1    = (const float*)d_in[6];
  const float* g1    = (const float*)d_in[7];
  const float* beta1 = (const float*)d_in[8];
  const float* W2    = (const float*)d_in[9];
  const float* b2    = (const float*)d_in[10];
  const float* g2    = (const float*)d_in[11];
  const float* beta2 = (const float*)d_in[12];
  const float* eps   = (const float*)d_in[13];
  float* out = (float*)d_out;

  // Workspace layout
  char* ws = (char*)d_ws;
  size_t off = 0;
  auto take = [&](size_t bytes) { char* p = ws + off; off += (bytes + 255) & ~(size_t)255; return p; };
  float*          aggr  = (float*)         take((size_t)cN * cH * 4);      // scatter target
  unsigned short* hbf   = (unsigned short*)take((size_t)cN * cH * 2);      // bf16 of (1+eps)x+aggr
  float*          h1    = (float*)         take((size_t)cN * 2 * cH * 4);  // GEMM1 out (pre-BN)
  unsigned short* h1a   = (unsigned short*)take((size_t)cN * 2 * cH * 2);  // bf16 silu(bn(h1))
  float*          h2    = (float*)         take((size_t)cN * cH * 4);      // GEMM2 out (pre-BN)
  unsigned short* WeT   = (unsigned short*)take((size_t)cH * cED * 2);     // [512][64]  bf16
  unsigned short* W1T   = (unsigned short*)take((size_t)2 * cH * cH * 2);  // [1024][512] bf16
  unsigned short* W2T   = (unsigned short*)take((size_t)cH * 2 * cH * 2);  // [512][1024] bf16
  float* colsum1 = (float*)take(2 * cH * 4);
  float* colsq1  = (float*)take(2 * cH * 4);
  float* colsum2 = (float*)take(cH * 4);
  float* colsq2  = (float*)take(cH * 4);
  float* mean1   = (float*)take(2 * cH * 4);
  float* rstd1   = (float*)take(2 * cH * 4);
  float* mean2   = (float*)take(cH * 4);
  float* rstd2   = (float*)take(cH * 4);

  const long long nNH  = (long long)cN * cH;       // 25.6M
  const long long nN2H = (long long)cN * 2 * cH;   // 51.2M

  // 0) zero accumulators (must happen every call for graph replay)
  k_zero<<<2048, 256, 0, stream>>>(aggr, nNH);
  k_zero<<<16, 256, 0, stream>>>(colsum1, 2 * cH);
  k_zero<<<16, 256, 0, stream>>>(colsq1, 2 * cH);
  k_zero<<<8, 256, 0, stream>>>(colsum2, cH);
  k_zero<<<8, 256, 0, stream>>>(colsq2, cH);

  // 1) weight conversion (f32 -> transposed bf16)
  k_convT<<<128, 256, 0, stream>>>(W_e, WeT, cED, cH);        // [64,512] -> [512][64]
  k_convT<<<2048, 256, 0, stream>>>(W1, W1T, cH, 2 * cH);     // [512,1024] -> [1024][512]
  k_convT<<<2048, 256, 0, stream>>>(W2, W2T, 2 * cH, cH);     // [1024,512] -> [512][1024]

  // 2) edge message + scatter (WMMA edge GEMM fused with gather/relu/atomics)
  k_edge<<<cE / 16, 256, 0, stream>>>(x, ea, ei, WeT, b_e, aggr);

  // 3) GIN combine -> bf16
  k_combine<<<4096, 256, 0, stream>>>(x, aggr, eps, hbf, nNH);

  // 4) GEMM1: h1 = hbf @ W1 + b1 (+ fused BN column stats); TDM-staged B tile
  {
    dim3 grid((3125 + 3) / 4, (2 * cH) / 128);   // 782 x 8
    const size_t sh = (size_t)128 * cH * 2;      // 128 cols x K=512 bf16 = 128 KB
    k_gemm_bias_stats<<<grid, 256, sh, stream>>>(hbf, W1T, b1, h1, colsum1, colsq1,
                                                 cN, cH, 2 * cH);
  }
  k_bnstats<<<(2 * cH + 255) / 256, 256, 0, stream>>>(colsum1, colsq1, mean1, rstd1,
                                                      2 * cH, 1.0f / (float)cN);
  // 5) BN + SiLU -> bf16
  k_bnact_bf16<<<8192, 256, 0, stream>>>(h1, mean1, rstd1, g1, beta1, h1a, nN2H, 2 * cH);

  // 6) GEMM2: h2 = h1a @ W2 + b2 (+ fused BN column stats); TDM-staged B tile
  {
    dim3 grid((3125 + 3) / 4, cH / 128);         // 782 x 4
    const size_t sh = (size_t)128 * 2 * cH * 2;  // 128 cols x K=1024 bf16 = 256 KB
    k_gemm_bias_stats<<<grid, 256, sh, stream>>>(h1a, W2T, b2, h2, colsum2, colsq2,
                                                 cN, 2 * cH, cH);
  }
  k_bnstats<<<(cH + 255) / 256, 256, 0, stream>>>(colsum2, colsq2, mean2, rstd2,
                                                  cH, 1.0f / (float)cN);
  // 7) outer BN + SiLU -> output (f32)
  k_bnact_f32<<<4096, 256, 0, stream>>>(h2, mean2, rstd2, g2, beta2, out, nNH, cH);
}
